// EquivariantTransformer_6244882448733
// MI455X (gfx1250) — matrix operand order repaired and checked by reading, not compiled, not run on hardware
//
#include <hip/hip_runtime.h>
#include <hip/hip_bf16.h>
#include <stdint.h>

// Problem constants (fixed by the reference)
#define BB   2
#define NN   2048
#define CC   256
#define HH   8
#define DHD  32
#define MCK  256
#define ROWS (BB * NN)     // 4096

typedef __attribute__((ext_vector_type(16))) _Float16 v16h;
typedef __attribute__((ext_vector_type(8)))  _Float16 v8h;
typedef __attribute__((ext_vector_type(8)))  float    v8f;
typedef __attribute__((vector_size(16)))     int      v4i;

#if __has_builtin(__builtin_amdgcn_global_load_async_to_lds_b128) && \
    __has_builtin(__builtin_amdgcn_s_wait_asynccnt)
#define USE_ASYNC_LDS 1
#else
#define USE_ASYNC_LDS 0
#endif

__device__ __forceinline__ float swishf(float x) {
    return x / (1.0f + __expf(-x));
}

// ---------------------------------------------------------------------------
// Weight transpose + f32->f16 convert: Wt[n][k] = (f16) W[k][n], 256x256.
// LDS-tiled so both global read and write are coalesced. One-time per launch.
// ---------------------------------------------------------------------------
__global__ __launch_bounds__(256)
void wtrans_kernel(const float* __restrict__ W, _Float16* __restrict__ Wt) {
    __shared__ float tile[32][33];
    const int tx = threadIdx.x;        // 0..31
    const int ty = threadIdx.y;        // 0..7
    const int x0 = blockIdx.x * 32;
    const int y0 = blockIdx.y * 32;
    #pragma unroll
    for (int j = 0; j < 4; ++j)
        tile[ty + j * 8][tx] = W[(size_t)(y0 + ty + j * 8) * CC + x0 + tx];
    __syncthreads();
    #pragma unroll
    for (int j = 0; j < 4; ++j)
        Wt[(size_t)(x0 + ty + j * 8) * CC + y0 + tx] =
            (_Float16)tile[tx][ty + j * 8];
}

// ---------------------------------------------------------------------------
// LayerNorm: one block (256 threads = C) per row. Wave32 shfl + LDS partials.
// Optionally emits f32 and/or f16 outputs (f16 feeds the WMMA GEMMs).
// ---------------------------------------------------------------------------
__global__ __launch_bounds__(256)
void layernorm_kernel(const float* __restrict__ x,
                      const float* __restrict__ g,
                      const float* __restrict__ b,
                      float* __restrict__ outf,        // may be nullptr
                      _Float16* __restrict__ outh) {   // may be nullptr
    const int row  = blockIdx.x;
    const int tid  = threadIdx.x;
    const int lane = tid & 31;
    const int wv   = tid >> 5;

    float v = x[(size_t)row * CC + tid];
    float s1 = v, s2 = v * v;
    #pragma unroll
    for (int m = 16; m > 0; m >>= 1) {
        s1 += __shfl_xor(s1, m, 32);
        s2 += __shfl_xor(s2, m, 32);
    }
    __shared__ float p1[8], p2[8], mv[2];
    if (lane == 0) { p1[wv] = s1; p2[wv] = s2; }
    __syncthreads();
    if (tid == 0) {
        float a = 0.f, c = 0.f;
        #pragma unroll
        for (int i = 0; i < 8; ++i) { a += p1[i]; c += p2[i]; }
        float mean = a / (float)CC;
        float var  = c / (float)CC - mean * mean;
        mv[0] = mean;
        mv[1] = rsqrtf(var + 1e-5f);
    }
    __syncthreads();
    float val = (v - mv[0]) * mv[1] * g[tid] + b[tid];
    if (outf) outf[(size_t)row * CC + tid] = val;
    if (outh) outh[(size_t)row * CC + tid] = (_Float16)val;
}

// ---------------------------------------------------------------------------
// Distance + exact top-256-smallest per row via full bitonic sort of
// (dist_bits << 32 | index) keys in LDS. dists >= 0 so float-bit order works;
// index in LSBs reproduces jax.lax.top_k's lowest-index tie-break.
// One block (256 threads) per (b, n) row.
// ---------------------------------------------------------------------------
__global__ __launch_bounds__(256)
void dist_topk_kernel(const float* __restrict__ pairwise_g,
                      const unsigned char* __restrict__ mask,
                      int* __restrict__ idx_out) {
    __shared__ unsigned long long keys[NN];   // 16 KB

    const int row = blockIdx.x;               // b*N + n
    const int b   = row >> 11;
    const int tid = threadIdx.x;

    const size_t base = ((size_t)row) * NN * 3;  // pairwise_g[b, n, :, :]
    #pragma unroll
    for (int t = 0; t < NN / 256; ++t) {
        int m = tid + t * 256;
        const float* gp = pairwise_g + base + (size_t)m * 3;
        float g0 = gp[0], g1 = gp[1], g2 = gp[2];
        float d = sqrtf(g0 * g0 + g1 * g1 + g2 * g2);
        if (mask[(size_t)b * NN + m] == 0) d = 1e8f;
        keys[m] = (((unsigned long long)__float_as_uint(d)) << 32) |
                  (unsigned long long)(unsigned)m;
    }
    __syncthreads();

    // Bitonic sort, ascending
    for (unsigned k2 = 2; k2 <= NN; k2 <<= 1) {
        for (unsigned j = k2 >> 1; j > 0; j >>= 1) {
            #pragma unroll
            for (int t = 0; t < NN / 256; ++t) {
                unsigned i   = (unsigned)tid + ((unsigned)t << 8);
                unsigned ixj = i ^ j;
                if (ixj > i) {
                    bool up = ((i & k2) == 0);
                    unsigned long long a = keys[i], c = keys[ixj];
                    if ((a > c) == up) { keys[i] = c; keys[ixj] = a; }
                }
            }
            __syncthreads();
        }
    }
    idx_out[(size_t)row * MCK + tid] = (int)(keys[tid] & 0xffffffffu);
}

// ---------------------------------------------------------------------------
// WMMA GEMM: out(M,256) = A(M,256) @ W(256,256) + bias [+ swish] [+ res]
// A: f16 row-major. Wt: f16 column-major (Wt[n][k]). Block = 256 threads
// (8 waves), tile 128x64, TK=32, K loop fully unrolled. Tile staging uses
// gfx1250 GLOBAL_LOAD_ASYNC_TO_LDS_B128 (ASYNCcnt) when available; else
// 16B vector copies. Fragments are two contiguous 16B runs (ds_load_b128).
// ---------------------------------------------------------------------------
#define GK  256
#define GN  256
#define ALD 40
#define BLD 40

__global__ __launch_bounds__(256)
void gemm_wmma_kernel(const _Float16* __restrict__ A,
                      const _Float16* __restrict__ Wt,
                      const float* __restrict__ bias,
                      const float* __restrict__ res,     // may be nullptr
                      float* __restrict__ outf,          // may be nullptr
                      _Float16* __restrict__ outh,       // may be nullptr
                      int act) {
    __shared__ _Float16 Alds[128 * ALD];   // A[r][k], stride 40 (80B, 16B-aligned)
    __shared__ _Float16 Blds[64 * BLD];    // B[c][k], stride 40

    const int tid   = threadIdx.x;
    const int lane  = tid & 31;
    const int wv    = tid >> 5;        // wave 0..7 -> 16-row strip
    const int hi    = lane >> 4;       // lane half-group
    const int l16   = lane & 15;
    const int mbase = blockIdx.x * 128;
    const int nbase = blockIdx.y * 64;

    // Per-thread staging coordinates (constant across K steps)
    const int rA0 = (tid)       >> 2, cA0 = (tid)       & 3;   // A chunk 0
    const int rA1 = (tid + 256) >> 2, cA1 = (tid + 256) & 3;   // A chunk 1
    const int cB  = tid >> 2,        kB  = tid & 3;            // B chunk

    _Float16* lA0 = &Alds[rA0 * ALD + cA0 * 8];
    _Float16* lA1 = &Alds[rA1 * ALD + cA1 * 8];
    _Float16* lB  = &Blds[cB * BLD + kB * 8];

    v8f acc[4];
    #pragma unroll
    for (int s = 0; s < 4; ++s)
        #pragma unroll
        for (int e = 0; e < 8; ++e) acc[s][e] = 0.0f;

    #pragma unroll
    for (int kb = 0; kb < GK; kb += 32) {
        __syncthreads();
        const _Float16* gA0 = &A[(size_t)(mbase + rA0) * GK + kb + cA0 * 8];
        const _Float16* gA1 = &A[(size_t)(mbase + rA1) * GK + kb + cA1 * 8];
        const _Float16* gB  = &Wt[(size_t)(nbase + cB) * GK + kb + kB * 8];
#if USE_ASYNC_LDS
        // Direct memory->LDS, no VGPR round-trip; tracked by ASYNCcnt
        __builtin_amdgcn_global_load_async_to_lds_b128((v4i*)gA0, (v4i*)lA0, 0, 0);
        __builtin_amdgcn_global_load_async_to_lds_b128((v4i*)gA1, (v4i*)lA1, 0, 0);
        __builtin_amdgcn_global_load_async_to_lds_b128((v4i*)gB,  (v4i*)lB,  0, 0);
        __builtin_amdgcn_s_wait_asynccnt(0);
#else
        *(v8h*)lA0 = *(const v8h*)gA0;
        *(v8h*)lA1 = *(const v8h*)gA1;
        *(v8h*)lB  = *(const v8h*)gB;
#endif
        __syncthreads();

        // A fragment: two contiguous 8-half runs per lane
        const _Float16* ap = &Alds[(wv * 16 + l16) * ALD + hi * 8];
        v8h a0 = *(const v8h*)(ap);
        v8h a1 = *(const v8h*)(ap + 16);
        v16h af = __builtin_shufflevector(a0, a1, 0, 1, 2, 3, 4, 5, 6, 7,
                                                  8, 9, 10, 11, 12, 13, 14, 15);
        #pragma unroll
        for (int sub = 0; sub < 4; ++sub) {
            const _Float16* bp = &Blds[(sub * 16 + l16) * BLD + hi * 8];
            v8h b0 = *(const v8h*)(bp);
            v8h b1 = *(const v8h*)(bp + 16);
            v16h bf = __builtin_shufflevector(b0, b1, 0, 1, 2, 3, 4, 5, 6, 7,
                                                      8, 9, 10, 11, 12, 13, 14, 15);
            acc[sub] = __builtin_amdgcn_wmma_f32_16x16x32_f16(
                false, af, false, bf, (short)0, acc[sub], false, false);
        }
    }

    // Epilogue: C/D layout -> row = v + 8*hi, col = lane%16
    #pragma unroll
    for (int sub = 0; sub < 4; ++sub) {
        int col = nbase + sub * 16 + l16;
        float bv = bias[col];
        #pragma unroll
        for (int v = 0; v < 8; ++v) {
            int row = mbase + wv * 16 + v + hi * 8;
            float val = acc[sub][v] + bv;
            if (act == 1) val = swishf(val);
            if (res) val += res[(size_t)row * GN + col];
            if (outf) outf[(size_t)row * GN + col] = val;
            if (outh) outh[(size_t)row * GN + col] = (_Float16)val;
        }
    }
}

// ---------------------------------------------------------------------------
// Fused neighbourhood attention: WeightNet MLP + gathered QK scores +
// per-head wave32 softmax + gathered V aggregation. One block per (b,n) row.
// Emits f16 (feeds the output-projection WMMA GEMM).
// ---------------------------------------------------------------------------
__global__ __launch_bounds__(256)
void attention_kernel(const float* __restrict__ pairwise_g,
                      const unsigned char* __restrict__ mask,
                      const int* __restrict__ idx,
                      const float* __restrict__ qbuf,
                      const float* __restrict__ kbuf,
                      const float* __restrict__ vbuf,
                      const float* __restrict__ wn_w1, const float* __restrict__ wn_b1,
                      const float* __restrict__ wn_w2, const float* __restrict__ wn_b2,
                      const float* __restrict__ wn_w3, const float* __restrict__ wn_b3,
                      _Float16* __restrict__ ema_h) {
    __shared__ float q_s[CC];
    __shared__ int   idx_s[MCK];
    __shared__ float pre[MCK * HH];          // scores -> attn, 8 KB
    __shared__ float w1[3 * 16], b1[16];
    __shared__ float w2[16 * 16], b2[16];
    __shared__ float w3[16 * 8],  b3[8];

    const int row  = blockIdx.x;             // b*N + n
    const int b    = row >> 11;
    const int tid  = threadIdx.x;
    const int lane = tid & 31;
    const int wv   = tid >> 5;

    q_s[tid]   = qbuf[(size_t)row * CC + tid];
    idx_s[tid] = idx[(size_t)row * MCK + tid];
    if (tid < 48)  w1[tid] = wn_w1[tid];
    if (tid < 16)  b1[tid] = wn_b1[tid];
    if (tid >= 64 && tid < 64 + 256) w2[tid - 64] = wn_w2[tid - 64];
    if (tid < 16)  b2[tid] = wn_b2[tid];
    if (tid >= 96 && tid < 96 + 128) w3[tid - 96] = wn_w3[tid - 96];
    if (tid < 8)   b3[tid] = wn_b3[tid];
    __syncthreads();

    // --- Phase 1: per-neighbor WeightNet + QK scores (thread t = neighbor t)
    {
        const int m = idx_s[tid];
        const float* gp = pairwise_g + (((size_t)row) * NN + m) * 3;
        float g0 = gp[0], g1 = gp[1], g2 = gp[2];

        float h1[16];
        #pragma unroll
        for (int k = 0; k < 16; ++k)
            h1[k] = swishf(g0 * w1[k] + g1 * w1[16 + k] + g2 * w1[32 + k] + b1[k]);
        float h2[16];
        #pragma unroll
        for (int k = 0; k < 16; ++k) {
            float a = b2[k];
            #pragma unroll
            for (int j = 0; j < 16; ++j) a += h1[j] * w2[j * 16 + k];
            h2[k] = swishf(a);
        }
        const bool mk = mask[(size_t)b * NN + m] != 0;
        const float* krow = kbuf + (size_t)((size_t)b * NN + m) * CC;
        const float inv_sqrt_dh = 0.17677669529663687f;  // 1/sqrt(32)
        #pragma unroll
        for (int h = 0; h < HH; ++h) {
            float loc = b3[h];
            #pragma unroll
            for (int j = 0; j < 16; ++j) loc += h2[j] * w3[j * 8 + h];
            float s = 0.f;
            #pragma unroll 8
            for (int d = 0; d < DHD; ++d)
                s += q_s[h * DHD + d] * krow[h * DHD + d];
            pre[tid * HH + h] = mk ? (loc + s * inv_sqrt_dh) : -1e38f;
        }
    }
    __syncthreads();

    // --- Phase 2: softmax over neighbors, head h = wave id, wave32 butterfly
    {
        const int h = wv;
        float mx = -3.4e38f;
        #pragma unroll
        for (int i = 0; i < 8; ++i)
            mx = fmaxf(mx, pre[(lane + i * 32) * HH + h]);
        #pragma unroll
        for (int m = 16; m > 0; m >>= 1) mx = fmaxf(mx, __shfl_xor(mx, m, 32));
        float sm = 0.f;
        #pragma unroll
        for (int i = 0; i < 8; ++i)
            sm += __expf(pre[(lane + i * 32) * HH + h] - mx);
        #pragma unroll
        for (int m = 16; m > 0; m >>= 1) sm += __shfl_xor(sm, m, 32);
        float inv = 1.0f / sm;
        #pragma unroll
        for (int i = 0; i < 8; ++i) {
            int t = lane + i * 32;
            pre[t * HH + h] = __expf(pre[t * HH + h] - mx) * inv;
        }
    }
    __syncthreads();

    // --- Phase 3: aggregate gathered V. thread -> (h = tid/32, d = tid%32)
    {
        const int h = tid >> 5;
        const int d = tid & 31;
        float acc = 0.f;
        for (int k = 0; k < MCK; ++k) {
            const int m = idx_s[k];
            acc += pre[k * HH + h] *
                   vbuf[(size_t)((size_t)b * NN + m) * CC + h * DHD + d];
        }
        ema_h[(size_t)row * CC + h * DHD + d] = (_Float16)acc;
    }
}

// ---------------------------------------------------------------------------
// Launcher
// ---------------------------------------------------------------------------
extern "C" void kernel_launch(void* const* d_in, const int* in_sizes, int n_in,
                              void* d_out, int out_size, void* d_ws, size_t ws_size,
                              hipStream_t stream) {
    (void)in_sizes; (void)n_in; (void)out_size; (void)ws_size;

    const float*         pairwise = (const float*)d_in[0];
    const float*         x        = (const float*)d_in[1];
    const unsigned char* mask     = (const unsigned char*)d_in[2];
    const float* ln1_g  = (const float*)d_in[3];
    const float* ln1_b  = (const float*)d_in[4];
    const float* ln2_g  = (const float*)d_in[5];
    const float* ln2_b  = (const float*)d_in[6];
    const float* wn_w1  = (const float*)d_in[7];
    const float* wn_b1  = (const float*)d_in[8];
    const float* wn_w2  = (const float*)d_in[9];
    const float* wn_b2  = (const float*)d_in[10];
    const float* wn_w3  = (const float*)d_in[11];
    const float* wn_b3  = (const float*)d_in[12];
    const float* wq     = (const float*)d_in[13];
    const float* bq     = (const float*)d_in[14];
    const float* wk     = (const float*)d_in[15];
    const float* bk     = (const float*)d_in[16];
    const float* in_w   = (const float*)d_in[17];
    const float* in_b   = (const float*)d_in[18];
    const float* out_w  = (const float*)d_in[19];
    const float* out_b  = (const float*)d_in[20];
    const float* mlp_w1 = (const float*)d_in[21];
    const float* mlp_b1 = (const float*)d_in[22];
    const float* mlp_w2 = (const float*)d_in[23];
    const float* mlp_b2 = (const float*)d_in[24];

    float* outp = (float*)d_out;

    const size_t RC = (size_t)ROWS * CC;     // 1,048,576 elems per (rows x C) buf
    const size_t WSZ = (size_t)CC * CC;      // 65,536 elems per weight matrix

    float* ws = (float*)d_ws;
    float* qb = ws + 0 * RC;                 // f32 (rows x C)
    float* kb = ws + 1 * RC;
    float* vb = ws + 2 * RC;
    float* x1 = ws + 3 * RC;
    _Float16* hbase   = (_Float16*)(ws + 4 * RC);
    _Float16* x_ln_h  = hbase + 0 * RC;      // f16 (rows x C)
    _Float16* ema_h   = hbase + 1 * RC;
    _Float16* hb_h    = hbase + 2 * RC;
    _Float16* t1_h    = hbase + 3 * RC;
    _Float16* wq_t    = hbase + 4 * RC + 0 * WSZ;   // f16 transposed weights
    _Float16* wk_t    = hbase + 4 * RC + 1 * WSZ;
    _Float16* in_w_t  = hbase + 4 * RC + 2 * WSZ;
    _Float16* out_w_t = hbase + 4 * RC + 3 * WSZ;
    _Float16* m_w1_t  = hbase + 4 * RC + 4 * WSZ;
    _Float16* m_w2_t  = hbase + 4 * RC + 5 * WSZ;
    int*      idxb    = (int*)(hbase + 4 * RC + 6 * WSZ);  // rows x MC ints

    const dim3 blk(256);
    const dim3 grid_rows(ROWS);
    const dim3 grid_gemm(ROWS / 128, CC / 64);
    const dim3 blk_t(32, 8);
    const dim3 grid_t(CC / 32, CC / 32);

    // 0) Transpose+convert the six weight matrices to f16 column-major
    hipLaunchKernelGGL(wtrans_kernel, grid_t, blk_t, 0, stream, wq,     wq_t);
    hipLaunchKernelGGL(wtrans_kernel, grid_t, blk_t, 0, stream, wk,     wk_t);
    hipLaunchKernelGGL(wtrans_kernel, grid_t, blk_t, 0, stream, in_w,   in_w_t);
    hipLaunchKernelGGL(wtrans_kernel, grid_t, blk_t, 0, stream, out_w,  out_w_t);
    hipLaunchKernelGGL(wtrans_kernel, grid_t, blk_t, 0, stream, mlp_w1, m_w1_t);
    hipLaunchKernelGGL(wtrans_kernel, grid_t, blk_t, 0, stream, mlp_w2, m_w2_t);

    // 1) LN1 (f16 copy feeds the q/k/v GEMMs)
    hipLaunchKernelGGL(layernorm_kernel, grid_rows, blk, 0, stream,
                       x, ln1_g, ln1_b, (float*)nullptr, x_ln_h);
    // 2) Q, K, V projections (WMMA)
    hipLaunchKernelGGL(gemm_wmma_kernel, grid_gemm, blk, 0, stream,
                       x_ln_h, wq_t,   bq,   (const float*)nullptr, qb, (_Float16*)nullptr, 0);
    hipLaunchKernelGGL(gemm_wmma_kernel, grid_gemm, blk, 0, stream,
                       x_ln_h, wk_t,   bk,   (const float*)nullptr, kb, (_Float16*)nullptr, 0);
    hipLaunchKernelGGL(gemm_wmma_kernel, grid_gemm, blk, 0, stream,
                       x_ln_h, in_w_t, in_b, (const float*)nullptr, vb, (_Float16*)nullptr, 0);
    // 3) distances + exact top-256
    hipLaunchKernelGGL(dist_topk_kernel, grid_rows, blk, 0, stream, pairwise, mask, idxb);
    // 4) fused WeightNet + attention + aggregation (emits f16)
    hipLaunchKernelGGL(attention_kernel, grid_rows, blk, 0, stream,
                       pairwise, mask, idxb, qb, kb, vb,
                       wn_w1, wn_b1, wn_w2, wn_b2, wn_w3, wn_b3, ema_h);
    // 5) output projection + residual with original x -> x1 (f32)
    hipLaunchKernelGGL(gemm_wmma_kernel, grid_gemm, blk, 0, stream,
                       ema_h, out_w_t, out_b, x, x1, (_Float16*)nullptr, 0);
    // 6) LN2 (f16 only; feeds MLP GEMM)
    hipLaunchKernelGGL(layernorm_kernel, grid_rows, blk, 0, stream,
                       x1, ln2_g, ln2_b, (float*)nullptr, hb_h);
    // 7) MLP: swish(h @ w1 + b1) (f16) then @ w2 + b2 + x1 -> d_out (f32)
    hipLaunchKernelGGL(gemm_wmma_kernel, grid_gemm, blk, 0, stream,
                       hb_h, m_w1_t, mlp_b1, (const float*)nullptr,
                       (float*)nullptr, t1_h, 1);
    hipLaunchKernelGGL(gemm_wmma_kernel, grid_gemm, blk, 0, stream,
                       t1_h, m_w2_t, mlp_b2, x1, outp, (_Float16*)nullptr, 0);
}
